// SparseMoELayer_55362128446067
// MI455X (gfx1250) — matrix operand order repaired and checked by reading, not compiled, not run on hardware
//
#include <hip/hip_runtime.h>

// ---------------- problem constants ----------------
constexpr int Bc = 4, Tc = 2048, Dc = 1024, Ec = 8, Kc = 2, Hc = 2048;
constexpr int Nc    = Bc * Tc;      // 8192 tokens
constexpr int SLOTS = Nc * Kc;      // 16384 (token, expert) slots total
constexpr int TM = 128, TN = 128, KS = 32;   // GEMM tile sizes

// ---------------- CDNA5 async global->LDS support (guarded) ----------------
#if defined(__has_builtin)
#if __has_builtin(__builtin_amdgcn_global_load_async_to_lds_b128)
#define HAVE_ASYNC_LDS 1
#endif
#if __has_builtin(__builtin_amdgcn_s_wait_asynccnt)
#define HAVE_WAIT_ASYNC 1
#endif
#endif

typedef int vint4 __attribute__((vector_size(16)));  // GCC vector: matches builtin proto

#ifdef HAVE_ASYNC_LDS
// proto: arg0 = global src (vint4 AS1*), arg1 = LDS dst (vint4 AS3*), imm offset, cpol
#define ASYNC_CP16(dst, src)                                                   \
  __builtin_amdgcn_global_load_async_to_lds_b128(                              \
      (__attribute__((address_space(1))) vint4*)(src),                         \
      (__attribute__((address_space(3))) vint4*)(dst), 0, 0)
#else
#define ASYNC_CP16(dst, src) (*(uint4*)(dst) = *(const uint4*)(src))
#endif

static __device__ __forceinline__ void async_join() {
#ifdef HAVE_ASYNC_LDS
#ifdef HAVE_WAIT_ASYNC
  __builtin_amdgcn_s_wait_asynccnt(0);
#else
  asm volatile("s_wait_asynccnt 0x0" ::: "memory");
#endif
#endif
  __syncthreads();
}

// ---------------- WMMA types ----------------
typedef __attribute__((ext_vector_type(16))) __bf16 v16bf;
typedef __attribute__((ext_vector_type(8)))  float  v8f;

union Frag16 {         // 16 bf16 = 32 bytes, loaded as two 16B chunks
  uint4 q[2];
  v16bf v;
};

static __device__ __forceinline__ unsigned short f32_to_bf16(float f) {
  unsigned int u = __float_as_uint(f);
  u += 0x7FFFu + ((u >> 16) & 1u);        // round-to-nearest-even
  return (unsigned short)(u >> 16);
}

// ---------------- elementwise kernels ----------------
__global__ __launch_bounds__(256) void zero_kernel(float* out, int n, int* cnt) {
  int i = blockIdx.x * 256 + threadIdx.x;
  if (i < n) out[i] = 0.0f;
  if (i < Ec) cnt[i] = 0;
}

// converts pairs of f32 -> packed 2x bf16 (one u32 store)
__global__ __launch_bounds__(256) void cvt_bf16_kernel(const float* __restrict__ src,
                                                       unsigned int* __restrict__ dst,
                                                       int npairs) {
  int i = blockIdx.x * 256 + threadIdx.x;
  if (i < npairs) {
    float2 f = ((const float2*)src)[i];
    unsigned int lo = f32_to_bf16(f.x);
    unsigned int hi = f32_to_bf16(f.y);
    dst[i] = lo | (hi << 16);
  }
}

// ---------------- router: one wave per token ----------------
__global__ __launch_bounds__(256) void router_kernel(const float* __restrict__ x,
                                                     const float* __restrict__ gate_w,
                                                     int* __restrict__ cnt,
                                                     int* __restrict__ tok_ids,
                                                     float* __restrict__ tok_w) {
  const int wave = threadIdx.x >> 5;
  const int lane = threadIdx.x & 31;
  const int t = blockIdx.x * 8 + wave;
  if (t >= Nc) return;

  float s[Ec];
#pragma unroll
  for (int e = 0; e < Ec; ++e) s[e] = 0.0f;

  for (int d = lane; d < Dc; d += 32) {
    float xv = x[(size_t)t * Dc + d];
#pragma unroll
    for (int e = 0; e < Ec; ++e) s[e] += xv * gate_w[e * Dc + d];
  }
#pragma unroll
  for (int e = 0; e < Ec; ++e)
    for (int off = 16; off > 0; off >>= 1) s[e] += __shfl_xor(s[e], off, 32);

  if (lane == 0) {
    int e0 = 0;
    for (int e = 1; e < Ec; ++e) if (s[e] > s[e0]) e0 = e;
    int e1 = (e0 == 0) ? 1 : 0;
    for (int e = 0; e < Ec; ++e) if (e != e0 && s[e] > s[e1]) e1 = e;
    // softmax over the two selected logits (max first, like top_k order)
    float z1 = __expf(s[e1] - s[e0]);
    float inv = 1.0f / (1.0f + z1);
    int p0 = atomicAdd(&cnt[e0], 1);
    tok_ids[e0 * Nc + p0] = t;
    tok_w[e0 * Nc + p0] = inv;
    int p1 = atomicAdd(&cnt[e1], 1);
    tok_ids[e1 * Nc + p1] = t;
    tok_w[e1 * Nc + p1] = z1 * inv;
  }
}

__global__ void scan_kernel(const int* __restrict__ cnt, int* __restrict__ offs) {
  if (threadIdx.x == 0 && blockIdx.x == 0) {
    int a = 0;
    for (int e = 0; e < Ec; ++e) { offs[e] = a; a += cnt[e]; }
  }
}

// ---------------- FFN stage 1: h = silu(x@Wg^T) * (x@Wi^T) ----------------
// grid: (nt = H/TN, mt = N/TM, e)
__global__ __launch_bounds__(256) void ffn1_kernel(const unsigned short* __restrict__ xb,   // [N,D] bf16
                                                   const unsigned short* __restrict__ wg,   // [E,H,D] bf16
                                                   const unsigned short* __restrict__ wi,   // [E,H,D] bf16
                                                   const int* __restrict__ cnt,
                                                   const int* __restrict__ offs,
                                                   const int* __restrict__ tok_ids,         // [E,N]
                                                   unsigned short* __restrict__ h_ws) {     // [SLOTS,H] bf16
  const int nt = blockIdx.x, mt = blockIdx.y, e = blockIdx.z;
  const int Me = cnt[e];
  if (mt * TM >= Me) return;
  const int offs_e = offs[e];

  __shared__ __align__(16) unsigned short As[2][TM * KS];  // 2 x 8 KB double-buffered token tile
  __shared__ int stok[TM];

  const int tid  = threadIdx.x;
  const int wave = tid >> 5;
  const int lane = tid & 31;

  if (tid < TM) {
    int slot = mt * TM + tid;
    if (slot >= Me) slot = Me - 1;          // padding rows clamp (results unused)
    stok[tid] = tok_ids[e * Nc + slot];
  }
  __syncthreads();

  // each thread stages exactly two 16B chunks per K-step; row/part fixed per thread
  const int row0 = tid >> 2, part = (tid & 3) * 8;       // chunk 0: c = tid
  const int row1 = row0 + 64;                            // chunk 1: c = tid + 256
  const unsigned short* sA0 = xb + (size_t)stok[row0] * Dc + part;
  const unsigned short* sA1 = xb + (size_t)stok[row1] * Dc + part;
  const int dA0 = row0 * KS + part;
  const int dA1 = row1 * KS + part;

  auto stage = [&](int buf, int kb) {
    ASYNC_CP16(&As[buf][dA0], sA0 + kb);
    ASYNC_CP16(&As[buf][dA1], sA1 + kb);
  };

  v8f accg[8] = {};
  v8f acci[8] = {};

  const int nbase = nt * TN + wave * 16 + (lane & 15);
  const unsigned short* wgp = wg + ((size_t)e * Hc + nbase) * Dc + ((lane >> 4) * 16);
  const unsigned short* wip = wi + ((size_t)e * Hc + nbase) * Dc + ((lane >> 4) * 16);
  const int arow  = (lane & 15) * KS;
  const int akoff = (lane >> 4) * 16;

  Frag16 bg0, bi0, bg1, bi1;

  // prologue: stage tile 0 + load B for step 0
  stage(0, 0);
  bg0.q[0] = *(const uint4*)(wgp);
  bg0.q[1] = *(const uint4*)(wgp + 8);
  bi0.q[0] = *(const uint4*)(wip);
  bi0.q[1] = *(const uint4*)(wip + 8);
  async_join();

  // one K-step: prefetch next tile (A via async-LDS, B via pipelined regs), WMMA on current
  auto step = [&](int kb, int cur, Frag16& bgc, Frag16& bic,
                  int nxt, Frag16& bgn, Frag16& bin) {
    if (kb + KS < Dc) {
      stage(nxt, kb + KS);
      bgn.q[0] = *(const uint4*)(wgp + kb + KS);
      bgn.q[1] = *(const uint4*)(wgp + kb + KS + 8);
      bin.q[0] = *(const uint4*)(wip + kb + KS);
      bin.q[1] = *(const uint4*)(wip + kb + KS + 8);
    }
    Frag16 a[8];
#pragma unroll
    for (int f = 0; f < 8; ++f) {
      const unsigned short* ap = &As[cur][f * 16 * KS + arow];
      a[f].q[0] = *(const uint4*)(ap + akoff);
      a[f].q[1] = *(const uint4*)(ap + 8 + akoff);
    }
#pragma unroll
    for (int f = 0; f < 8; ++f) {
      accg[f] = __builtin_amdgcn_wmma_f32_16x16x32_bf16(false, a[f].v, false, bgc.v,
                                                        (short)0, accg[f], false, false);
      acci[f] = __builtin_amdgcn_wmma_f32_16x16x32_bf16(false, a[f].v, false, bic.v,
                                                        (short)0, acci[f], false, false);
    }
    async_join();
  };

  for (int kb = 0; kb < Dc; kb += 2 * KS) {
    step(kb,      0, bg0, bi0, 1, bg1, bi1);
    step(kb + KS, 1, bg1, bi1, 0, bg0, bi0);
  }

  // h = silu(g) * i, store bf16 into compacted slot rows
  const int ncol = nt * TN + wave * 16 + (lane & 15);
#pragma unroll
  for (int f = 0; f < 8; ++f) {
#pragma unroll
    for (int v = 0; v < 8; ++v) {
      int m = f * 16 + (lane >> 4) * 8 + v;
      int gslot = mt * TM + m;
      if (gslot < Me) {
        float g = accg[f][v];
        float hval = (g / (1.0f + __expf(-g))) * acci[f][v];
        h_ws[(size_t)(offs_e + gslot) * Hc + ncol] = f32_to_bf16(hval);
      }
    }
  }
}

// ---------------- FFN stage 2: out += w * (h @ Wo^T) ----------------
// grid: (nt = D/TN, mt = N/TM, e)
__global__ __launch_bounds__(256) void ffn2_kernel(const unsigned short* __restrict__ h_ws, // [SLOTS,H] bf16
                                                   const unsigned short* __restrict__ wo,   // [E,D,H] bf16
                                                   const int* __restrict__ cnt,
                                                   const int* __restrict__ offs,
                                                   const int* __restrict__ tok_ids,
                                                   const float* __restrict__ tok_w,
                                                   float* __restrict__ out) {
  const int nt = blockIdx.x, mt = blockIdx.y, e = blockIdx.z;
  const int Me = cnt[e];
  if (mt * TM >= Me) return;
  const int offs_e = offs[e];

  __shared__ __align__(16) unsigned short As[2][TM * KS];
  __shared__ int   stok[TM];
  __shared__ float stw[TM];

  const int tid  = threadIdx.x;
  const int wave = tid >> 5;
  const int lane = tid & 31;

  if (tid < TM) {
    int slot = mt * TM + tid;
    if (slot >= Me) slot = Me - 1;
    stok[tid] = tok_ids[e * Nc + slot];
    stw[tid]  = tok_w[e * Nc + slot];
  }
  __syncthreads();

  const int row0 = tid >> 2, part = (tid & 3) * 8;
  const int row1 = row0 + 64;
  int gr0 = mt * TM + row0; if (gr0 >= Me) gr0 = Me - 1;
  int gr1 = mt * TM + row1; if (gr1 >= Me) gr1 = Me - 1;
  const unsigned short* sA0 = h_ws + (size_t)(offs_e + gr0) * Hc + part;
  const unsigned short* sA1 = h_ws + (size_t)(offs_e + gr1) * Hc + part;
  const int dA0 = row0 * KS + part;
  const int dA1 = row1 * KS + part;

  auto stage = [&](int buf, int kb) {
    ASYNC_CP16(&As[buf][dA0], sA0 + kb);
    ASYNC_CP16(&As[buf][dA1], sA1 + kb);
  };

  v8f acc[8] = {};

  const int nbase = nt * TN + wave * 16 + (lane & 15);
  const unsigned short* wop = wo + ((size_t)e * Dc + nbase) * Hc + ((lane >> 4) * 16);
  const int arow  = (lane & 15) * KS;
  const int akoff = (lane >> 4) * 16;

  Frag16 b0, b1;

  stage(0, 0);
  b0.q[0] = *(const uint4*)(wop);
  b0.q[1] = *(const uint4*)(wop + 8);
  async_join();

  auto step = [&](int kb, int cur, Frag16& bc, int nxt, Frag16& bn) {
    if (kb + KS < Hc) {
      stage(nxt, kb + KS);
      bn.q[0] = *(const uint4*)(wop + kb + KS);
      bn.q[1] = *(const uint4*)(wop + kb + KS + 8);
    }
    Frag16 a[8];
#pragma unroll
    for (int f = 0; f < 8; ++f) {
      const unsigned short* ap = &As[cur][f * 16 * KS + arow];
      a[f].q[0] = *(const uint4*)(ap + akoff);
      a[f].q[1] = *(const uint4*)(ap + 8 + akoff);
    }
#pragma unroll
    for (int f = 0; f < 8; ++f) {
      acc[f] = __builtin_amdgcn_wmma_f32_16x16x32_bf16(false, a[f].v, false, bc.v,
                                                       (short)0, acc[f], false, false);
    }
    async_join();
  };

  for (int kb = 0; kb < Hc; kb += 2 * KS) {
    step(kb,      0, b0, 1, b1);
    step(kb + KS, 1, b1, 0, b0);
  }

  // scale by routing weight and scatter-add (exactly 2 commutative adds/element -> deterministic)
  const int ncol = nt * TN + wave * 16 + (lane & 15);
#pragma unroll
  for (int f = 0; f < 8; ++f) {
#pragma unroll
    for (int v = 0; v < 8; ++v) {
      int m = f * 16 + (lane >> 4) * 8 + v;
      if (mt * TM + m < Me) {
        float val = acc[f][v] * stw[m];
        atomicAdd(&out[(size_t)stok[m] * Dc + ncol], val);
      }
    }
  }
}

// ---------------- host-side launch ----------------
extern "C" void kernel_launch(void* const* d_in, const int* in_sizes, int n_in,
                              void* d_out, int out_size, void* d_ws, size_t ws_size,
                              hipStream_t stream) {
  const float* x      = (const float*)d_in[0];  // [B,T,D]
  const float* gate_w = (const float*)d_in[1];  // [E,D]
  const float* w_gate = (const float*)d_in[2];  // [E,H,D]
  const float* w_in   = (const float*)d_in[3];  // [E,H,D]
  const float* w_out  = (const float*)d_in[4];  // [E,D,H]
  float* out = (float*)d_out;                   // [B,T,D]

  char* ws = (char*)d_ws;
  size_t off = 0;
  auto carve = [&](size_t bytes) -> void* {
    void* p = ws + off;
    off = (off + bytes + 255) & ~(size_t)255;
    return p;
  };
  unsigned short* xb   = (unsigned short*)carve((size_t)Nc * Dc * 2);        // 16 MB
  unsigned short* wgb  = (unsigned short*)carve((size_t)Ec * Hc * Dc * 2);   // 32 MB
  unsigned short* wib  = (unsigned short*)carve((size_t)Ec * Hc * Dc * 2);   // 32 MB
  unsigned short* wob  = (unsigned short*)carve((size_t)Ec * Dc * Hc * 2);   // 32 MB
  int*            cnt  = (int*)carve(Ec * sizeof(int));
  int*            offs = (int*)carve(Ec * sizeof(int));
  int*            tids = (int*)carve((size_t)Ec * Nc * sizeof(int));         // 256 KB
  float*          tw   = (float*)carve((size_t)Ec * Nc * sizeof(float));     // 256 KB
  unsigned short* h_ws = (unsigned short*)carve((size_t)SLOTS * Hc * 2);     // 64 MB

  // 1) zero output + expert counts
  {
    int n = Nc * Dc;
    zero_kernel<<<(n + 255) / 256, 256, 0, stream>>>(out, n, cnt);
  }
  // 2) f32 -> bf16 conversions (weights become L2-resident at 100 MB total)
  {
    int p = (Nc * Dc) / 2;
    cvt_bf16_kernel<<<(p + 255) / 256, 256, 0, stream>>>(x, (unsigned int*)xb, p);
    p = (Ec * Hc * Dc) / 2;
    cvt_bf16_kernel<<<(p + 255) / 256, 256, 0, stream>>>(w_gate, (unsigned int*)wgb, p);
    cvt_bf16_kernel<<<(p + 255) / 256, 256, 0, stream>>>(w_in,  (unsigned int*)wib, p);
    cvt_bf16_kernel<<<(p + 255) / 256, 256, 0, stream>>>(w_out, (unsigned int*)wob, p);
  }
  // 3) router: top-2 + softmax + per-expert token gather
  router_kernel<<<Nc / 8, 256, 0, stream>>>(x, gate_w, cnt, tids, tw);
  // 4) exclusive scan of the 8 counts
  scan_kernel<<<1, 32, 0, stream>>>(cnt, offs);
  // 5) h = silu(x Wg^T) * (x Wi^T) for the gathered tokens
  ffn1_kernel<<<dim3(Hc / TN, Nc / TM, Ec), 256, 0, stream>>>(xb, wgb, wib, cnt, offs, tids, h_ws);
  // 6) out += routing_w * (h Wo^T)
  ffn2_kernel<<<dim3(Dc / TN, Nc / TM, Ec), 256, 0, stream>>>(h_ws, wob, cnt, offs, tids, tw, out);
}
